// Wav2VecContBertLoss_89601607729733
// MI455X (gfx1250) — matrix operand order repaired and checked by reading, not compiled, not run on hardware
//
#include <hip/hip_runtime.h>
#include <stdint.h>

// Problem constants from the reference (fixed shapes).
#define B_    8
#define C_    256
#define T_    1024
#define M_    512
#define K_NEG 100

typedef __attribute__((ext_vector_type(16))) _Float16 v16h;
typedef __attribute__((ext_vector_type(8)))  _Float16 v8h;
typedef __attribute__((ext_vector_type(8)))  float    v8f;

// ---------------------------------------------------------------------------
// threefry2x32-20, matching JAX's implementation (key = {0, 42} for key(42)).
// ---------------------------------------------------------------------------
__device__ __forceinline__ uint32_t rotl32(uint32_t x, int r) {
  return (x << r) | (x >> (32 - r));
}

__device__ __forceinline__ void threefry2x32(uint32_t k0, uint32_t k1,
                                             uint32_t x0, uint32_t x1,
                                             uint32_t& o0, uint32_t& o1) {
  uint32_t ks[3];
  ks[0] = k0; ks[1] = k1; ks[2] = k0 ^ k1 ^ 0x1BD11BDAu;
  const int R0[4] = {13, 15, 26, 6};
  const int R1[4] = {17, 29, 16, 24};
  x0 += ks[0]; x1 += ks[1];
#pragma unroll
  for (int g = 0; g < 5; ++g) {
    const int* R = (g & 1) ? R1 : R0;
#pragma unroll
    for (int r = 0; r < 4; ++r) { x0 += x1; x1 = rotl32(x1, R[r]); x1 ^= x0; }
    x0 += ks[(g + 1) % 3];
    x1 += ks[(g + 2) % 3] + (uint32_t)(g + 1);
  }
  o0 = x0; o1 = x1;
}

// ---------------------------------------------------------------------------
// Kernel 1: per-batch stable scan of mask -> masked time indices; zero d_out.
// ---------------------------------------------------------------------------
__global__ void build_idx_kernel(const int* __restrict__ mask,
                                 int* __restrict__ idx,
                                 float* __restrict__ out) {
  int b = threadIdx.x;
  if (b == 0) { out[0] = 0.0f; out[1] = 0.0f; }
  if (b < B_) {
    const int* mrow = mask + b * T_;
    int* irow = idx + b * M_;
    int cnt = 0;
    for (int t = 0; t < T_; ++t) {
      if (mrow[t] != 0) { if (cnt < M_) irow[cnt] = t; ++cnt; }
    }
  }
}

// ---------------------------------------------------------------------------
// Kernel 2: gather masked columns, transpose to [B, M, C], convert to f16.
//   mf[b,m,c] = feat_proj[b,c,idx[b,m]],  mc[b,m,c] = context[b,c,idx[b,m]]
// One block per (b, c); threads stride over m (gathers stay inside one 4KB row).
// ---------------------------------------------------------------------------
__global__ void gather_f16_kernel(const float* __restrict__ feat,
                                  const float* __restrict__ ctx,
                                  const int* __restrict__ idx,
                                  _Float16* __restrict__ mfh,
                                  _Float16* __restrict__ mch) {
  int blk = blockIdx.x;
  int b = blk / C_;
  int c = blk % C_;
  const float* frow = feat + ((size_t)b * C_ + c) * T_;
  const float* crow = ctx  + ((size_t)b * C_ + c) * T_;
  const int* irow = idx + b * M_;
  for (int m = threadIdx.x; m < M_; m += blockDim.x) {
    int t = irow[m];
    size_t o = ((size_t)b * M_ + m) * C_ + c;
    mfh[o] = (_Float16)frow[t];
    mch[o] = (_Float16)crow[t];
  }
}

// ---------------------------------------------------------------------------
// Kernel 3: batched Gram matrix G[b] = mc[b] (512x256) @ mf[b]^T via WMMA.
// One wave per 16x16 tile; 8 K-steps of v_wmma_f32_16x16x32_f16.
// Fragment layouts per CDNA5 ISA 7.12.2 (16-bit A 16x32, B 32x16, f32 C/D).
// ---------------------------------------------------------------------------
__global__ void gram_wmma_kernel(const _Float16* __restrict__ mch,
                                 const _Float16* __restrict__ mfh,
                                 float* __restrict__ G) {
  int wid  = blockIdx.x * (blockDim.x >> 5) + (threadIdx.x >> 5);
  int lane = threadIdx.x & 31;
  int b    = wid >> 10;              // 1024 tiles (32x32) per batch
  int rem  = wid & 1023;
  int tm   = (rem >> 5) << 4;        // tile row base in M
  int tn   = (rem & 31) << 4;        // tile col base in N
  int laneHi = lane >> 4;            // 0: lanes 0-15, 1: lanes 16-31
  int lane15 = lane & 15;

  // A row for this lane: mc[b, tm+lane15, :]; B "column" n: mf[b, tn+lane15, :]
  const _Float16* Abase = mch + ((size_t)b * M_ + (tm + lane15)) * C_;
  const _Float16* Bbase = mfh + ((size_t)b * M_ + (tn + lane15)) * C_;

  v8f acc = {};
#pragma unroll
  for (int k0 = 0; k0 < C_; k0 += 32) {
    // A 16x32 f16: lane<16 holds K {0..7, 16..23}; lane>=16 holds {8..15, 24..31}
    int ha = k0 + (laneHi ? 8 : 0);
    v8h alo = *(const v8h*)(Abase + ha);
    v8h ahi = *(const v8h*)(Abase + ha + 16);
    v16h a;
#pragma unroll
    for (int i = 0; i < 8; ++i) { a[i] = alo[i]; a[8 + i] = ahi[i]; }
    // B 32x16 f16: lane<16 holds K {0..7, 16..23} of col n; lane>=16 holds {8..15, 24..31}
    int hb = k0 + (laneHi ? 8 : 0);
    v8h blo = *(const v8h*)(Bbase + hb);
    v8h bhi = *(const v8h*)(Bbase + hb + 16);
    v16h bf;
#pragma unroll
    for (int i = 0; i < 8; ++i) { bf[i] = blo[i]; bf[8 + i] = bhi[i]; }
    acc = __builtin_amdgcn_wmma_f32_16x16x32_f16(
        /*neg_a=*/false, a, /*neg_b=*/false, bf,
        /*c_mod=*/(short)0, acc, /*reuse_a=*/false, /*reuse_b=*/false);
  }

  // D 16x16 f32: VGPR v, lanes 0-15 -> M = v, lanes 16-31 -> M = v + 8.
  float* Gt = G + (size_t)b * M_ * M_;
#pragma unroll
  for (int v = 0; v < 8; ++v) {
    int row = tm + v + (laneHi ? 8 : 0);
    Gt[(size_t)row * M_ + tn + lane15] = acc[v];
  }
}

// ---------------------------------------------------------------------------
// Kernel 4: per (b,m) row -- regenerate threefry scores, top-100 negatives,
// softmax prob of positive + clamped-logit aux loss; atomic-accumulate.
// ---------------------------------------------------------------------------
__global__ void loss_kernel(const float* __restrict__ G,
                            float* __restrict__ out) {
  __shared__ float sc[M_];
  __shared__ float lg[K_NEG + 1];
  __shared__ unsigned long long redu[128];
  __shared__ float redf[128];
  __shared__ float redf2[128];

  int tid = threadIdx.x;
  int b = blockIdx.x >> 9;
  int m = blockIdx.x & (M_ - 1);

  // JAX counter-mode uniform: N = B*M*M, out[p<H] = o0(p, p+H); out[p>=H] = o1(p-H, p)
  const uint32_t H = (uint32_t)B_ * M_ * M_ / 2u;  // 1,048,576
  uint32_t base = (uint32_t)(((b * M_) + m) * M_);
  for (int j = tid; j < M_; j += blockDim.x) {
    uint32_t p = base + (uint32_t)j;
    uint32_t x0 = (p < H) ? p : (p - H);
    uint32_t x1 = (p < H) ? (p + H) : p;
    uint32_t o0, o1;
    threefry2x32(0u, 42u, x0, x1, o0, o1);
    uint32_t bits = (p < H) ? o0 : o1;
    float f = __uint_as_float((bits >> 9) | 0x3f800000u) - 1.0f;
    sc[j] = (j == m) ? -1.0f : f;   // self-mask
  }
  __syncthreads();

  const float* Grow = G + ((size_t)b * M_ + m) * M_;

  // top-100 by iterative argmax (smaller index wins ties, like lax.top_k)
  for (int k = 0; k < K_NEG; ++k) {
    unsigned long long best = 0ull;
    for (int j = tid; j < M_; j += blockDim.x) {
      uint32_t u = __float_as_uint(sc[j]);
      uint32_t key = (u & 0x80000000u) ? ~u : (u | 0x80000000u);  // order-preserving
      unsigned long long pk =
          ((unsigned long long)key << 32) | (uint32_t)(M_ - 1 - j);
      if (pk > best) best = pk;
    }
    redu[tid] = best;
    __syncthreads();
    for (int s = blockDim.x >> 1; s > 0; s >>= 1) {
      if (tid < s) { unsigned long long o = redu[tid + s]; if (o > redu[tid]) redu[tid] = o; }
      __syncthreads();
    }
    int jstar = (M_ - 1) - (int)(redu[0] & 0xFFFFFFFFu);
    if (tid == 0) { lg[k + 1] = Grow[jstar]; sc[jstar] = -2.0f; }
    __syncthreads();
  }
  if (tid == 0) lg[0] = Grow[m];   // positive logit = diagonal of Gram
  __syncthreads();

  // softmax max
  float lm = -1e30f;
  for (int i = tid; i < K_NEG + 1; i += blockDim.x) lm = fmaxf(lm, lg[i]);
  redf[tid] = lm;
  __syncthreads();
  for (int s = blockDim.x >> 1; s > 0; s >>= 1) {
    if (tid < s) redf[tid] = fmaxf(redf[tid], redf[tid + s]);
    __syncthreads();
  }
  float mx = redf[0];
  __syncthreads();

  // sum(exp) and sum(tanh^2) in one pass (CLAMP_W == 1)
  float sexp = 0.0f, saux = 0.0f;
  for (int i = tid; i < K_NEG + 1; i += blockDim.x) {
    float l = lg[i];
    sexp += expf(l - mx);
    float th = tanhf(l);
    saux += th * th;
  }
  redf[tid] = sexp; redf2[tid] = saux;
  __syncthreads();
  for (int s = blockDim.x >> 1; s > 0; s >>= 1) {
    if (tid < s) { redf[tid] += redf[tid + s]; redf2[tid] += redf2[tid + s]; }
    __syncthreads();
  }
  if (tid == 0) {
    float p0 = expf(lg[0] - mx) / redf[0];
    atomicAdd(&out[0], p0);
    atomicAdd(&out[1], redf2[0]);
  }
}

// ---------------------------------------------------------------------------
extern "C" void kernel_launch(void* const* d_in, const int* in_sizes, int n_in,
                              void* d_out, int out_size, void* d_ws, size_t ws_size,
                              hipStream_t stream) {
  (void)in_sizes; (void)n_in; (void)out_size; (void)ws_size;
  const float* feat = (const float*)d_in[0];   // [B, C, T] f32
  const int*   mask = (const int*)d_in[1];     // [B, T] bool -> int
  const float* ctx  = (const float*)d_in[2];   // [B, C, T] f32
  float* out = (float*)d_out;                  // {total_loss, total_aux_loss}

  // Workspace carve-out (~12.6 MB): idx | mf_f16 | mc_f16 | G
  char* w = (char*)d_ws;
  int* idx = (int*)w;                 w += (size_t)B_ * M_ * sizeof(int);
  _Float16* mfh = (_Float16*)w;       w += (size_t)B_ * M_ * C_ * sizeof(_Float16);
  _Float16* mch = (_Float16*)w;       w += (size_t)B_ * M_ * C_ * sizeof(_Float16);
  float* G = (float*)w;

  build_idx_kernel<<<1, 32, 0, stream>>>(mask, idx, out);
  gather_f16_kernel<<<B_ * C_, 256, 0, stream>>>(feat, ctx, idx, mfh, mch);
  // 8192 waves = B * 32 * 32 tiles; 8 waves / 256-thread block
  gram_wmma_kernel<<<1024, 256, 0, stream>>>(mch, mfh, G);
  loss_kernel<<<B_ * M_, 128, 0, stream>>>(G, out);
}
